// MultiHeadSelfAttention_58841051955504
// MI455X (gfx1250) — matrix operand order repaired and checked by reading, not compiled, not run on hardware
//
#include <hip/hip_runtime.h>
#include <hip/hip_bf16.h>
#include <math.h>

// ---------------------------------------------------------------------------
// MHSA forward for MI455X (gfx1250, wave32, WMMA bf16 w/ f32 accumulate).
// x:[2,2048,1024] f32 -> out:[2,2048,1024] f32
// Pipeline: qkv_gemm (fused bias + head-transpose, bf16 out)
//           -> flash attention (fused QK^T / online softmax / PV, bf16 out,
//              K tiles staged by the Tensor Data Mover)
//           -> proj_gemm (A tiles staged by TDM, fused bias, f32 out)
// ---------------------------------------------------------------------------

typedef __bf16 bf16;
typedef __attribute__((ext_vector_type(16))) __bf16 v16bf;
typedef __attribute__((ext_vector_type(8)))  __bf16 bf16x8;
typedef __attribute__((ext_vector_type(8)))  float  v8f;
typedef __attribute__((ext_vector_type(4)))  unsigned int u32x4;
typedef __attribute__((ext_vector_type(8)))  int i32x8;
typedef __attribute__((ext_vector_type(4)))  int i32x4;

#define EMBED 1024
#define NHEAD 16
#define HDIM  64
#define SEQ   2048
#define BATCH 2
#define ROWS  (BATCH*SEQ)   // 4096
#define QKVN  (3*EMBED)     // 3072

static __device__ inline v16bf combine8(bf16x8 lo, bf16x8 hi) {
  v16bf r;
#pragma unroll
  for (int i = 0; i < 8; ++i) { r[i] = lo[i]; r[i + 8] = hi[i]; }
  return r;
}

static __device__ inline v8f wmma_bf16(v16bf a, v16bf b, v8f c) {
  // v_wmma_f32_16x16x32_bf16: D = A(16x32) * B(32x16) + C(16x16 f32)
  return __builtin_amdgcn_wmma_f32_16x16x32_bf16(false, a, false, b,
                                                 (short)0, c, false, false);
}

static __device__ inline bf16x8 cvt8(const float* src) {
  float4 f0 = *(const float4*)src;
  float4 f1 = *(const float4*)(src + 4);
  bf16x8 h;
  h[0] = (bf16)f0.x; h[1] = (bf16)f0.y; h[2] = (bf16)f0.z; h[3] = (bf16)f0.w;
  h[4] = (bf16)f1.x; h[5] = (bf16)f1.y; h[6] = (bf16)f1.z; h[7] = (bf16)f1.w;
  return h;
}

// ---------------------------------------------------------------------------
// Tensor Data Mover: async 2D bf16 tile load, global -> LDS.
// D# per CDNA5 ISA:
//   group0: [1:0]=count=1, [63:32]=lds_addr, [120:64]=global_addr, [127:126]=2
//   group1: [17:16]=data_size(1 => 2B), tensor_dim0/1, tile_dim0/1,
//           tensor_dim0_stride (all in element units)
// ---------------------------------------------------------------------------
static __device__ inline void tdm_load_2d_bf16(unsigned lds_addr,
                                               const bf16* gptr,
                                               unsigned tensor_d0, unsigned tensor_d1,
                                               unsigned tile_d0, unsigned tile_d1,
                                               unsigned stride_d0) {
  unsigned long long ga = (unsigned long long)(size_t)gptr;
  u32x4 g0;
  g0[0] = 1u;                                        // count=1, user descriptor
  g0[1] = lds_addr;                                  // LDS byte address
  g0[2] = (unsigned)(ga & 0xffffffffu);              // global_addr[31:0]
  g0[3] = (unsigned)((ga >> 32) & 0x01ffffffu)       // global_addr[56:32]
        | (2u << 30);                                // type=2 ("image")
  i32x8 g1;
  g1[0] = (int)(1u << 16);                           // wg_mask=0, data_size=2B
  g1[1] = (int)((tensor_d0 & 0xffffu) << 16);        // [63:48]=dim0[15:0]
  g1[2] = (int)((tensor_d0 >> 16) | ((tensor_d1 & 0xffffu) << 16));
  g1[3] = (int)((tensor_d1 >> 16) | (tile_d0 << 16)); // [127:112]=tile_dim0
  g1[4] = (int)(tile_d1 & 0xffffu);                  // tile_dim1, tile_dim2=0
  g1[5] = (int)stride_d0;                            // dim0_stride[31:0]
  g1[6] = 0;                                         // stride bits, dim1_stride
  g1[7] = 0;
  i32x4 z4 = {0, 0, 0, 0};
#if defined(__clang_major__) && (__clang_major__ >= 23)
  i32x8 z8 = {0, 0, 0, 0, 0, 0, 0, 0};
  __builtin_amdgcn_tensor_load_to_lds(g0, g1, z4, z4, z8, 0);
#else
  __builtin_amdgcn_tensor_load_to_lds(g0, g1, z4, z4, 0);
#endif
}

// ---------------------------------------------------------------------------
// Kernel 1: qkv = x @ W_qkv + b_qkv, scattered head-major into q/k/v (bf16).
// Block tile 64(M) x 128(N), K-step 32; 8 waves, each 32x32 (2x2 WMMA tiles).
// ---------------------------------------------------------------------------
__global__ __launch_bounds__(256)
void qkv_gemm_kernel(const float* __restrict__ x, const float* __restrict__ Wqkv,
                     const float* __restrict__ bqkv,
                     bf16* __restrict__ qT, bf16* __restrict__ kT,
                     bf16* __restrict__ vT)
{
  __shared__ __align__(16) bf16 As[64 * 32];   // row-major [64][32]
  __shared__ __align__(16) bf16 Bt[128 * 32];  // transposed: Bt[j][k]

  const int tid  = threadIdx.x;
  const int wave = tid >> 5;
  const int lane = tid & 31;
  const int half = lane >> 4;
  const int l16  = lane & 15;
  const int rowBase = blockIdx.y * 64;
  const int colBase = blockIdx.x * 128;
  const int wm = (wave & 1) * 32;
  const int wn = (wave >> 1) * 32;

  v8f acc[2][2] = {};

  for (int k0 = 0; k0 < EMBED; k0 += 32) {
    // Stage A: 64x32 fp32 -> bf16, 8 packed elems/thread (b128 in, b128 out).
    {
      int r = tid >> 2, seg = (tid & 3) * 8;
      bf16x8 h = cvt8(&x[(size_t)(rowBase + r) * EMBED + k0 + seg]);
      *(bf16x8*)&As[r * 32 + seg] = h;
    }
    // Stage B transposed: Bt[j][k] = W[k0+k][colBase+j]; float4 reads.
    {
      int kk = tid >> 3, g = (tid & 7) * 16;
      const float* src = &Wqkv[(size_t)(k0 + kk) * QKVN + colBase + g];
#pragma unroll
      for (int i = 0; i < 16; i += 4) {
        float4 f = *(const float4*)(src + i);
        Bt[(g + i + 0) * 32 + kk] = (bf16)f.x;
        Bt[(g + i + 1) * 32 + kk] = (bf16)f.y;
        Bt[(g + i + 2) * 32 + kk] = (bf16)f.z;
        Bt[(g + i + 3) * 32 + kk] = (bf16)f.w;
      }
    }
    __syncthreads();

    v16bf afr[2], bfr[2];
#pragma unroll
    for (int i = 0; i < 2; ++i) {   // A frag: row=l16, K = half*8+{0..7} / +16
      int m = wm + i * 16 + l16;
      afr[i] = combine8(*(const bf16x8*)&As[m * 32 + half * 8],
                        *(const bf16x8*)&As[m * 32 + 16 + half * 8]);
    }
#pragma unroll
    for (int j = 0; j < 2; ++j) {   // B frag: col=l16, K = half*8+{0..7} / +16
      int n = wn + j * 16 + l16;
      bfr[j] = combine8(*(const bf16x8*)&Bt[n * 32 + half * 8],
                        *(const bf16x8*)&Bt[n * 32 + 16 + half * 8]);
    }
#pragma unroll
    for (int i = 0; i < 2; ++i)
#pragma unroll
      for (int j = 0; j < 2; ++j)
        acc[i][j] = wmma_bf16(afr[i], bfr[j], acc[i][j]);
    __syncthreads();
  }

  // Epilogue: +bias, scatter into [B,H,N,D] bf16 per component.
  const int mOff = half * 8;
#pragma unroll
  for (int i = 0; i < 2; ++i) {
#pragma unroll
    for (int j = 0; j < 2; ++j) {
      int jg = colBase + wn + j * 16 + l16;
      float bias = bqkv[jg];
      int comp = jg >> 10;          // 0=q 1=k 2=v
      int h = (jg & 1023) >> 6;
      int d = jg & 63;
      bf16* dst = comp == 0 ? qT : (comp == 1 ? kT : vT);
#pragma unroll
      for (int v = 0; v < 8; ++v) {
        int mg = rowBase + wm + i * 16 + mOff + v;
        int bb = mg >> 11;          // /SEQ
        int n = mg & 2047;
        dst[(((size_t)(bb * NHEAD + h) * SEQ) + n) * HDIM + d] =
            (bf16)(acc[i][j][v] + bias);
      }
    }
  }
}

// ---------------------------------------------------------------------------
// Kernel 2: flash attention per (b,h). 8 waves x 16 query rows = 128 rows/block.
// Key-step 32; K tile staged row-major by the TDM, V tile transposed by VALU.
// ---------------------------------------------------------------------------
__global__ __launch_bounds__(256)
void attn_kernel(const bf16* __restrict__ qT, const bf16* __restrict__ kT,
                 const bf16* __restrict__ vT, bf16* __restrict__ attnOut)
{
  __shared__ __align__(16) bf16 Kt[32 * 64];     // [key][feat], TDM target
  __shared__ __align__(16) bf16 Vt[64 * 32];     // [feat][key] (transposed)
  __shared__ __align__(16) bf16 Ps[8][16 * 32];  // per-wave P scratch

  const int tid  = threadIdx.x;
  const int wave = tid >> 5;
  const int lane = tid & 31;
  const int half = lane >> 4;
  const int l16  = lane & 15;

  const int bh   = blockIdx.x;           // b*NHEAD + h
  const int qblk = blockIdx.y;
  const size_t headBase = (size_t)bh * SEQ * HDIM;
  const int b = bh / NHEAD, h = bh % NHEAD;
  const int qRow0 = qblk * 128 + wave * 16;
  const unsigned ktLds = (unsigned)(size_t)&Kt[0];

  // Q fragments live in registers for the whole pass (2 K-steps of 32).
  v16bf qf[2];
#pragma unroll
  for (int s = 0; s < 2; ++s) {
    const bf16* qp = qT + headBase + (size_t)(qRow0 + l16) * HDIM + s * 32 + half * 8;
    qf[s] = combine8(*(const bf16x8*)qp, *(const bf16x8*)(qp + 16));
  }

  v8f o[4] = {};
  float mrun[8], lrun[8];
#pragma unroll
  for (int v = 0; v < 8; ++v) { mrun[v] = -1e30f; lrun[v] = 0.0f; }
  const float scale = 0.125f;            // HEAD_DIM^-0.5

  for (int kb = 0; kb < SEQ; kb += 32) {
    // Wave 0 kicks off the async K-tile DMA: 32x64 bf16, row-major.
    if (wave == 0) {
      tdm_load_2d_bf16(ktLds, kT + headBase + (size_t)kb * HDIM,
                       /*tensor*/ HDIM, SEQ, /*tile*/ HDIM, 32,
                       /*stride*/ HDIM);
    }
    // Meanwhile all threads stage V transposed (b128 global in, b16 scatter).
    {
      int r = tid >> 3, fseg = (tid & 7) * 8;
      bf16x8 vv = *(const bf16x8*)&vT[headBase + (size_t)(kb + r) * HDIM + fseg];
#pragma unroll
      for (int i = 0; i < 8; ++i) Vt[(fseg + i) * 32 + r] = vv[i];
    }
    if (kb + 32 < SEQ) {                 // pull next tiles toward L2/WGP$
      __builtin_prefetch(&kT[headBase + (size_t)(kb + 32 + (tid >> 3)) * HDIM], 0, 0);
      __builtin_prefetch(&vT[headBase + (size_t)(kb + 32 + (tid >> 3)) * HDIM], 0, 0);
    }
    if (wave == 0) __builtin_amdgcn_s_wait_tensorcnt(0);
    __syncthreads();

    // S(16x32) = Q(16x64) * K^T(64x32): two 16x16 tiles, 2 WMMA each.
    v8f sacc[2];
#pragma unroll
    for (int t = 0; t < 2; ++t) {
      v8f c = {};
#pragma unroll
      for (int s = 0; s < 2; ++s) {
        const bf16* bp = &Kt[(t * 16 + l16) * 64 + s * 32 + half * 8];
        v16bf bfr = combine8(*(const bf16x8*)bp, *(const bf16x8*)(bp + 16));
        c = wmma_bf16(qf[s], bfr, c);
      }
      sacc[t] = c;
    }

    // Online softmax: per accumulator slot v, row = half*8+v, cols across lanes.
    float alpha[8];
#pragma unroll
    for (int v = 0; v < 8; ++v) {
      float s0 = sacc[0][v] * scale;
      float s1 = sacc[1][v] * scale;
      float mx = fmaxf(s0, s1);
#pragma unroll
      for (int off = 1; off < 16; off <<= 1)
        mx = fmaxf(mx, __shfl_xor(mx, off, 16));
      float mnew = fmaxf(mrun[v], mx);
      float a  = __expf(mrun[v] - mnew);
      float p0 = __expf(s0 - mnew);
      float p1 = __expf(s1 - mnew);
      float rs = p0 + p1;
#pragma unroll
      for (int off = 1; off < 16; off <<= 1)
        rs += __shfl_xor(rs, off, 16);
      lrun[v] = lrun[v] * a + rs;
      mrun[v] = mnew;
      alpha[v] = a;
      int row = half * 8 + v;            // C-layout -> row-major P scratch
      Ps[wave][row * 32 + l16]      = (bf16)p0;
      Ps[wave][row * 32 + 16 + l16] = (bf16)p1;
    }
#pragma unroll
    for (int t = 0; t < 4; ++t)
#pragma unroll
      for (int v = 0; v < 8; ++v) o[t][v] *= alpha[v];

    // Wave-private LDS round trip (C-layout -> A-layout): wait our DS stores.
    asm volatile("s_wait_dscnt 0" ::: "memory");

    const bf16* pp = &Ps[wave][l16 * 32 + half * 8];
    v16bf pf = combine8(*(const bf16x8*)pp, *(const bf16x8*)(pp + 16));

    // O(16x64) += P(16x32) * V(32x64): 4 WMMA across feature tiles.
#pragma unroll
    for (int t = 0; t < 4; ++t) {
      const bf16* vp = &Vt[(t * 16 + l16) * 32 + half * 8];
      v16bf vfr = combine8(*(const bf16x8*)vp, *(const bf16x8*)(vp + 16));
      o[t] = wmma_bf16(pf, vfr, o[t]);
    }
    __syncthreads();
  }

  // Normalize and store into [B,N,C] bf16 (fused head-transpose).
#pragma unroll
  for (int v = 0; v < 8; ++v) {
    float inv = 1.0f / lrun[v];
    int n = qRow0 + half * 8 + v;
    size_t rowOff = ((size_t)(b * SEQ) + n) * EMBED + h * HDIM;
#pragma unroll
    for (int t = 0; t < 4; ++t)
      attnOut[rowOff + t * 16 + l16] = (bf16)(o[t][v] * inv);
  }
}

// ---------------------------------------------------------------------------
// Kernel 3: out = attn(bf16) @ W_proj + b_proj, fp32 output.
// A tiles (bf16, layout-preserving) are staged by the TDM.
// ---------------------------------------------------------------------------
__global__ __launch_bounds__(256)
void proj_gemm_kernel(const bf16* __restrict__ A, const float* __restrict__ Wp,
                      const float* __restrict__ bp, float* __restrict__ out)
{
  __shared__ __align__(16) bf16 As[64 * 32];   // TDM target
  __shared__ __align__(16) bf16 Bt[128 * 32];

  const int tid  = threadIdx.x;
  const int wave = tid >> 5;
  const int lane = tid & 31;
  const int half = lane >> 4;
  const int l16  = lane & 15;
  const int rowBase = blockIdx.y * 64;
  const int colBase = blockIdx.x * 128;
  const int wm = (wave & 1) * 32;
  const int wn = (wave >> 1) * 32;
  const unsigned asLds = (unsigned)(size_t)&As[0];

  v8f acc[2][2] = {};

  for (int k0 = 0; k0 < EMBED; k0 += 32) {
    // Wave 0: async DMA of the 64x32 bf16 A tile (row stride EMBED).
    if (wave == 0) {
      tdm_load_2d_bf16(asLds, A + (size_t)rowBase * EMBED + k0,
                       /*tensor*/ EMBED, ROWS, /*tile*/ 32, 64,
                       /*stride*/ EMBED);
    }
    // All threads stage B transposed: Bt[j][k] = Wp[k0+k][colBase+j].
    {
      int kk = tid >> 3, g = (tid & 7) * 16;
      const float* src = &Wp[(size_t)(k0 + kk) * EMBED + colBase + g];
#pragma unroll
      for (int i = 0; i < 16; i += 4) {
        float4 f = *(const float4*)(src + i);
        Bt[(g + i + 0) * 32 + kk] = (bf16)f.x;
        Bt[(g + i + 1) * 32 + kk] = (bf16)f.y;
        Bt[(g + i + 2) * 32 + kk] = (bf16)f.z;
        Bt[(g + i + 3) * 32 + kk] = (bf16)f.w;
      }
    }
    if (wave == 0) __builtin_amdgcn_s_wait_tensorcnt(0);
    __syncthreads();

    v16bf afr[2], bfr[2];
#pragma unroll
    for (int i = 0; i < 2; ++i) {
      int m = wm + i * 16 + l16;
      afr[i] = combine8(*(const bf16x8*)&As[m * 32 + half * 8],
                        *(const bf16x8*)&As[m * 32 + 16 + half * 8]);
    }
#pragma unroll
    for (int j = 0; j < 2; ++j) {
      int n = wn + j * 16 + l16;
      bfr[j] = combine8(*(const bf16x8*)&Bt[n * 32 + half * 8],
                        *(const bf16x8*)&Bt[n * 32 + 16 + half * 8]);
    }
#pragma unroll
    for (int i = 0; i < 2; ++i)
#pragma unroll
      for (int j = 0; j < 2; ++j)
        acc[i][j] = wmma_bf16(afr[i], bfr[j], acc[i][j]);
    __syncthreads();
  }

  const int mOff = half * 8;
#pragma unroll
  for (int i = 0; i < 2; ++i) {
#pragma unroll
    for (int j = 0; j < 2; ++j) {
      int jg = colBase + wn + j * 16 + l16;
      float bias = bp[jg];
#pragma unroll
      for (int v = 0; v < 8; ++v) {
        int mg = rowBase + wm + i * 16 + mOff + v;
        out[(size_t)mg * EMBED + jg] = acc[i][j][v] + bias;
      }
    }
  }
}

// ---------------------------------------------------------------------------
extern "C" void kernel_launch(void* const* d_in, const int* in_sizes, int n_in,
                              void* d_out, int out_size, void* d_ws, size_t ws_size,
                              hipStream_t stream) {
  (void)in_sizes; (void)n_in; (void)out_size; (void)ws_size;
  const float* x    = (const float*)d_in[0];
  const float* Wqkv = (const float*)d_in[1];
  const float* bqkv = (const float*)d_in[2];
  const float* Wp   = (const float*)d_in[3];
  const float* bp   = (const float*)d_in[4];
  float* out = (float*)d_out;

  // Workspace layout (bf16): q,k,v head-major [B,H,N,D] + attn [B,N,C] = 32 MB.
  const size_t per = (size_t)ROWS * EMBED;   // 4M elements each
  bf16* qT   = (bf16*)d_ws;
  bf16* kT   = qT + per;
  bf16* vT   = kT + per;
  bf16* attn = vT + per;

  dim3 blk(256);
  qkv_gemm_kernel<<<dim3(QKVN / 128, ROWS / 64), blk, 0, stream>>>(
      x, Wqkv, bqkv, qT, kT, vT);
  attn_kernel<<<dim3(BATCH * NHEAD, SEQ / 128), blk, 0, stream>>>(
      qT, kT, vT, attn);
  proj_gemm_kernel<<<dim3(EMBED / 128, ROWS / 64), blk, 0, stream>>>(
      attn, Wp, bp, out);
}
